// MogonetGCN_75823352643690
// MI455X (gfx1250) — compile-verified, matrix operand order checked by reading, not run on hardware
//
#include <hip/hip_runtime.h>
#include <hip/hip_bf16.h>

// ---------------------------------------------------------------------------
// MogonetGCN on MI455X (gfx1250).
// 3x [ bf16-WMMA GEMM -> edge SPMM (f32 atomics, L2-resident) -> bias+lrelu ].
// GEMM operands live in global memory in v_wmma_f32_16x16x32_bf16 fragment
// order (ISA 7.12.2); the GEMM K-loop is software-pipelined (double-buffered
// global_load_b128 -> v_wmma) and the schedule is pinned with
// sched_group_barrier so loads for step s+1 stay in flight behind step s's
// WMMA burst.
// ---------------------------------------------------------------------------

typedef __attribute__((ext_vector_type(16))) __bf16 v16bf;
typedef __attribute__((ext_vector_type(8)))  float  v8f;

#define GCN_N     50000
#define GCN_E     800000
#define GCN_INC   512
#define GCN_H1    128
#define GCN_H2    128
#define GCN_H3    64
#define GCN_SLOPE 0.25f

// Padded row count: 391 blocks x 128 rows (waves past N compute garbage that
// is never stored; reads stay inside the padded ws buffer).
#define GCN_NPAD  50048

#if defined(__has_builtin)
#if __has_builtin(__builtin_amdgcn_sched_group_barrier)
#define MG_SCHED_GROUP(mask, size) __builtin_amdgcn_sched_group_barrier((mask), (size), 0)
#endif
#endif
#ifndef MG_SCHED_GROUP
#define MG_SCHED_GROUP(mask, size)
#endif

// Fragment index helpers (16-bit A 16x32 / B 32x16 layout, wave32):
//   lane = m + 16*half      (m = row%16 for A, n%16 for B)
//   element e (0..15) within the lane's 16 contiguous bf16
//   k within the 32-wide K-step:  k = e + 8*half + 8*(e>>3)
// inverse (producer side):  half = (k>>3)&1 ;  e = (k&7) + 8*(k>>4)
__device__ __forceinline__ size_t packA_idx(int row, int k, int K) {
  int rb = row >> 4, m = row & 15;
  int s = k >> 5, kk = k & 31;
  int half = (kk >> 3) & 1;
  int e = (kk & 7) + 8 * (kk >> 4);
  int lane = m + 16 * half;
  return (((size_t)rb * (K >> 5) + s) * 32 + lane) * 16 + e;
}

// ---------------- producers: write A-operand in fragment order -------------
__global__ void mg_pack_x(const float* __restrict__ src,
                          __bf16* __restrict__ dst, int n, int K) {
  int i = blockIdx.x * blockDim.x + threadIdx.x;
  if (i >= n) return;
  int row = i / K, k = i % K;
  float v = __builtin_nontemporal_load(&src[i]);   // x is streamed exactly once
  dst[packA_idx(row, k, K)] = (__bf16)v;
}

// bias + leakyReLU, emitted in packed-A fragment order for the next GEMM
__global__ void mg_epilogue_pack(const float* __restrict__ agg,
                                 const float* __restrict__ bias,
                                 __bf16* __restrict__ dst, int n, int d) {
  int i = blockIdx.x * blockDim.x + threadIdx.x;
  if (i >= n) return;
  int row = i / d, k = i % d;
  float v = agg[i] + bias[k];
  v = (v > 0.0f) ? v : GCN_SLOPE * v;
  dst[packA_idx(row, k, d)] = (__bf16)v;
}

// final layer: fp32 row-major to d_out
__global__ void mg_epilogue_f32(const float* __restrict__ agg,
                                const float* __restrict__ bias,
                                float* __restrict__ out, int n, int d) {
  int i = blockIdx.x * blockDim.x + threadIdx.x;
  if (i >= n) return;
  float v = __builtin_nontemporal_load(&agg[i]) + bias[i % d];
  v = (v > 0.0f) ? v : GCN_SLOPE * v;
  __builtin_nontemporal_store(v, &out[i]);
}

// ---------------- one-time: pack W[K x OUT] into B-fragment order ----------
// layout: (((s*NT + nt)*32 + lane)*16 + e)
__global__ void mg_pack_B(const float* __restrict__ W,
                          __bf16* __restrict__ Bpk, int K, int OUT) {
  int i = blockIdx.x * blockDim.x + threadIdx.x;
  if (i >= K * OUT) return;
  int e = i & 15;
  int lane = (i >> 4) & 31;
  int rest = i >> 9;
  int NT = OUT >> 4;
  int nt = rest % NT, s = rest / NT;
  int half = lane >> 4;
  int n = nt * 16 + (lane & 15);
  int kk = e + 8 * half + 8 * (e >> 3);
  Bpk[i] = (__bf16)W[(size_t)(s * 32 + kk) * OUT + n];
}

// ------------------------------- zero fp32 ---------------------------------
__global__ void mg_zero(float* __restrict__ p, int n) {
  int i = blockIdx.x * blockDim.x + threadIdx.x;
  if (i < n) p[i] = 0.0f;
}

// ------------------------------ WMMA GEMM ----------------------------------
// Z[nrows x OUT] = A @ B, operands already in fragment order.
// 256 threads = 8 waves; wave w owns 16 output rows; NT=OUT/16 col tiles.
// Double-buffered K loop: fragments for step s+1 are in flight while the
// 8 (or 4) WMMAs of step s execute; sched_group_barrier pins that order.
template <int NT>
__global__ __launch_bounds__(256)
void mg_gemm_wmma(const __bf16* __restrict__ Apk, const __bf16* __restrict__ Bpk,
                  float* __restrict__ Z, int nrows, int K) {
  constexpr int OUT = NT * 16;
  const int lane = threadIdx.x & 31;
  const int wave = threadIdx.x >> 5;
  const int rb   = blockIdx.x * 8 + wave;   // 16-row block handled by this wave
  const int S    = K >> 5;                  // number of 32-wide K steps

  const v16bf* A16 = (const v16bf*)Apk + (size_t)rb * S * 32 + lane; // 32B frags
  const v16bf* B16 = (const v16bf*)Bpk + lane;

  v8f acc[NT];
  v8f zero = {};
#pragma unroll
  for (int nt = 0; nt < NT; ++nt) acc[nt] = zero;

  // ---- prologue: fragments for s = 0 ----
  v16bf a = A16[0];
  v16bf b[NT];
#pragma unroll
  for (int nt = 0; nt < NT; ++nt) b[nt] = B16[(size_t)nt * 32];

  for (int s = 0; s < S; ++s) {
    // issue next iteration's loads first (clamped on last iter; values unused)
    const int sn = (s + 1 < S) ? s + 1 : s;
    v16bf an = A16[(size_t)sn * 32];
    v16bf bn[NT];
#pragma unroll
    for (int nt = 0; nt < NT; ++nt)
      bn[nt] = B16[((size_t)sn * NT + nt) * 32];

    // NT back-to-back WMMAs cover the in-flight load latency
#pragma unroll
    for (int nt = 0; nt < NT; ++nt)
      acc[nt] = __builtin_amdgcn_wmma_f32_16x16x32_bf16(
          false, a, false, b[nt], (short)0, acc[nt], false, false);

    // Pin schedule: all (NT+1)*2 b128 reads for s+1 issue before the NT
    // WMMAs of s, so the WMMA-group wait only covers iteration s's loads.
    MG_SCHED_GROUP(0x020, (NT + 1) * 2);  // VMEM reads
    MG_SCHED_GROUP(0x008, NT);            // MFMA/WMMA

    a = an;
#pragma unroll
    for (int nt = 0; nt < NT; ++nt) b[nt] = bn[nt];
  }

  // C layout: VGPR v -> M=v (lanes 0-15), M=v+8 (lanes 16-31)
  const int half = lane >> 4, l15 = lane & 15;
#pragma unroll
  for (int nt = 0; nt < NT; ++nt) {
#pragma unroll
    for (int v = 0; v < 8; ++v) {
      int row = rb * 16 + v + 8 * half;
      if (row < nrows) Z[(size_t)row * OUT + nt * 16 + l15] = acc[nt][v];
    }
  }
}

// ------------------------------- SPMM --------------------------------------
// agg[dst[e], :] += w[e] * Z[src[e], :]   one wave32 per edge, float4 chunks.
__global__ __launch_bounds__(256)
void mg_spmm(const float* __restrict__ Z, const float* __restrict__ ew,
             const int* __restrict__ esrc, const int* __restrict__ edst,
             float* __restrict__ agg, int nedges, int d) {
  int e = blockIdx.x * 8 + (threadIdx.x >> 5);
  if (e >= nedges) return;
  int lane = threadIdx.x & 31;
  float w = ew[e];
  const float4* srow = (const float4*)&Z[(size_t)esrc[e] * d];
  float* trow = &agg[(size_t)edst[e] * d];
  for (int c4 = lane; c4 * 4 < d; c4 += 32) {      // d=128: 1 iter, d=64: lanes<16
    float4 v = srow[c4];
    int c = c4 * 4;
    atomicAdd(&trow[c + 0], w * v.x);
    atomicAdd(&trow[c + 1], w * v.y);
    atomicAdd(&trow[c + 2], w * v.z);
    atomicAdd(&trow[c + 3], w * v.w);
  }
}

// ---------------------------------------------------------------------------
extern "C" void kernel_launch(void* const* d_in, const int* in_sizes, int n_in,
                              void* d_out, int out_size, void* d_ws, size_t ws_size,
                              hipStream_t stream) {
  (void)in_sizes; (void)n_in; (void)out_size; (void)ws_size;

  const float* x    = (const float*)d_in[0];
  const float* ew   = (const float*)d_in[1];
  const float* W1   = (const float*)d_in[2];
  const float* b1   = (const float*)d_in[3];
  const float* W2   = (const float*)d_in[4];
  const float* b2   = (const float*)d_in[5];
  const float* W3   = (const float*)d_in[6];
  const float* b3   = (const float*)d_in[7];
  const int*   esrc = (const int*)d_in[8];
  const int*   edst = (const int*)d_in[9];
  float* out = (float*)d_out;

  // ---- workspace carve-up (all sizes are 256B multiples) ----
  char* ws = (char*)d_ws;
  size_t off = 0;
  __bf16* ApkA = (__bf16*)(ws + off); off += (size_t)GCN_NPAD * GCN_INC * 2; // 51.2MB
  __bf16* ApkB = (__bf16*)(ws + off); off += (size_t)GCN_NPAD * GCN_H1 * 2;  // 12.8MB
  __bf16* B1p  = (__bf16*)(ws + off); off += (size_t)GCN_INC * GCN_H1 * 2;
  __bf16* B2p  = (__bf16*)(ws + off); off += (size_t)GCN_H1 * GCN_H2 * 2;
  __bf16* B3p  = (__bf16*)(ws + off); off += (size_t)GCN_H2 * GCN_H3 * 2;
  float*  Zf   = (float*)(ws + off);  off += (size_t)GCN_N * GCN_H1 * 4;     // 25.6MB
  float*  agg  = (float*)(ws + off);  off += (size_t)GCN_N * GCN_H1 * 4;     // 25.6MB

  const int TPB = 256;
  const int gemmBlocks = GCN_NPAD / 128;      // 391
  const int spmmBlocks = (GCN_E + 7) / 8;     // one wave per edge

  // ---- one-time packing ----
  {
    int n = GCN_N * GCN_INC;
    mg_pack_x<<<(n + TPB - 1) / TPB, TPB, 0, stream>>>(x, ApkA, n, GCN_INC);
    n = GCN_INC * GCN_H1;
    mg_pack_B<<<(n + TPB - 1) / TPB, TPB, 0, stream>>>(W1, B1p, GCN_INC, GCN_H1);
    n = GCN_H1 * GCN_H2;
    mg_pack_B<<<(n + TPB - 1) / TPB, TPB, 0, stream>>>(W2, B2p, GCN_H1, GCN_H2);
    n = GCN_H2 * GCN_H3;
    mg_pack_B<<<(n + TPB - 1) / TPB, TPB, 0, stream>>>(W3, B3p, GCN_H2, GCN_H3);
  }

  // ================= Layer 1: 512 -> 128 =================
  mg_gemm_wmma<8><<<gemmBlocks, TPB, 0, stream>>>(ApkA, B1p, Zf, GCN_N, GCN_INC);
  {
    int n = GCN_N * GCN_H1;
    mg_zero<<<(n + TPB - 1) / TPB, TPB, 0, stream>>>(agg, n);
    mg_spmm<<<spmmBlocks, TPB, 0, stream>>>(Zf, ew, esrc, edst, agg, GCN_E, GCN_H1);
    mg_epilogue_pack<<<(n + TPB - 1) / TPB, TPB, 0, stream>>>(agg, b1, ApkB, n, GCN_H1);
  }

  // ================= Layer 2: 128 -> 128 =================
  mg_gemm_wmma<8><<<gemmBlocks, TPB, 0, stream>>>(ApkB, B2p, Zf, GCN_N, GCN_H1);
  {
    int n = GCN_N * GCN_H2;
    mg_zero<<<(n + TPB - 1) / TPB, TPB, 0, stream>>>(agg, n);
    mg_spmm<<<spmmBlocks, TPB, 0, stream>>>(Zf, ew, esrc, edst, agg, GCN_E, GCN_H2);
    mg_epilogue_pack<<<(n + TPB - 1) / TPB, TPB, 0, stream>>>(agg, b2, ApkB, n, GCN_H2);
  }

  // ================= Layer 3: 128 -> 64 ==================
  mg_gemm_wmma<4><<<gemmBlocks, TPB, 0, stream>>>(ApkB, B3p, Zf, GCN_N, GCN_H2);
  {
    int n = GCN_N * GCN_H3;
    mg_zero<<<(n + TPB - 1) / TPB, TPB, 0, stream>>>(agg, n);
    mg_spmm<<<spmmBlocks, TPB, 0, stream>>>(Zf, ew, esrc, edst, agg, GCN_E, GCN_H3);
    mg_epilogue_f32<<<(n + TPB - 1) / TPB, TPB, 0, stream>>>(agg, b3, out, n, GCN_H3);
  }
}
